// Convolution_80917183857194
// MI455X (gfx1250) — matrix operand order repaired and checked by reading, not compiled, not run on hardware
//
#include <hip/hip_runtime.h>
#include <hip/hip_bf16.h>
#include <math.h>

typedef __attribute__((ext_vector_type(16))) _Float16 v16h;
typedef __attribute__((ext_vector_type(8)))  float    v8f;

#define FDIM    32      // F
#define NPATHS  11
#define FEAT    288     // 9*F
#define HIDDEN  64
#define NB      8
#define RADCOLS 352     // NPATHS*F
#define EB      16      // edges per block

// W3J flat offsets (dims per path, row-major (a*dj+b)*dk+c)
// p:  (0,0,0)=1 @0, (0,1,1)=9 @1, (0,2,2)=25 @10, (1,0,1)=9 @35, (1,1,0)=9 @44,
//     (1,1,2)=45 @53, (1,2,1)=45 @98, (2,0,2)=25 @143, (2,1,1)=45 @168,
//     (2,2,0)=25 @213, (2,2,2)=125 @238  -> total 363

// ---------------------------------------------------------------------------
// Setup kernel: compute PATH_W-scaled real Wigner-3j tensors (double precision)
// ---------------------------------------------------------------------------
struct cdbl { double re, im; };
__device__ __forceinline__ cdbl cmul(cdbl a, cdbl b){ return {a.re*b.re - a.im*b.im, a.re*b.im + a.im*b.re}; }
__device__ __forceinline__ cdbl cadd(cdbl a, cdbl b){ return {a.re+b.re, a.im+b.im}; }
__device__ __forceinline__ cdbl cscl(cdbl a, double s){ return {a.re*s, a.im*s}; }
__device__ __forceinline__ cdbl cconj(cdbl a){ return {a.re, -a.im}; }

__device__ double factd(int n){ double r = 1.0; for (int i = 2; i <= n; ++i) r *= (double)i; return r; }

__device__ double su2_cg(int j1, int j2, int j3, int m1, int m2, int m3) {
  if (m1 + m2 != m3) return 0.0;
  double pref = sqrt((2.0*j3 + 1.0) * factd(j1+j2-j3) * factd(j1-j2+j3) * factd(-j1+j2+j3) / factd(j1+j2+j3+1));
  pref *= sqrt(factd(j3+m3)*factd(j3-m3)*factd(j1-m1)*factd(j1+m1)*factd(j2-m2)*factd(j2+m2));
  double s = 0.0;
  for (int k = 0; k <= j1 + j2 - j3; ++k) {
    int d0 = k, d1 = j1+j2-j3-k, d2 = j1-m1-k, d3 = j2+m2-k, d4 = j3-j2+m1+k, d5 = j3-j1-m2+k;
    if (d0 < 0 || d1 < 0 || d2 < 0 || d3 < 0 || d4 < 0 || d5 < 0) continue;
    double term = 1.0 / (factd(d0)*factd(d1)*factd(d2)*factd(d3)*factd(d4)*factd(d5));
    s += (k & 1) ? -term : term;
  }
  return pref * s;
}

__device__ void build_q(int l, cdbl q[5][5]) {
  for (int a = 0; a < 5; ++a) for (int b = 0; b < 5; ++b) q[a][b] = {0.0, 0.0};
  const double is2 = 0.70710678118654752440;
  for (int m = -l; m < 0; ++m) {
    q[l+m][l-m] = { is2, 0.0};   // l + |m|
    q[l+m][l+m] = { 0.0, -is2};  // l - |m|
  }
  q[l][l] = {1.0, 0.0};
  for (int m = 1; m <= l; ++m) {
    double sg = (m & 1) ? -1.0 : 1.0;
    q[l+m][l+m] = { sg*is2, 0.0};
    q[l+m][l-m] = { 0.0,  sg*is2};
  }
  cdbl ph;
  switch (l & 3) { case 0: ph = {1,0}; break; case 1: ph = {0,-1}; break;
                   case 2: ph = {-1,0}; break; default: ph = {0,1}; break; }
  for (int a = 0; a < 2*l+1; ++a) for (int b = 0; b < 2*l+1; ++b) q[a][b] = cmul(ph, q[a][b]);
}

__global__ void w3j_init_kernel(float* __restrict__ w3j_out) {
  if (threadIdx.x != 0 || blockIdx.x != 0) return;
  const int pi_[NPATHS] = {0,0,0,1,1,1,1,2,2,2,2};
  const int pj_[NPATHS] = {0,1,2,0,1,1,2,0,1,2,2};
  const int pk_[NPATHS] = {0,1,2,1,0,2,1,2,1,0,2};
  const double pwk[3] = { sqrt(1.0/3.0), sqrt(3.0/4.0), sqrt(5.0/4.0) }; // sqrt((2k+1)/cnt[k])
  int off = 0;
  for (int p = 0; p < NPATHS; ++p) {
    int l1 = pi_[p], l2 = pj_[p], l3 = pk_[p];
    int d1 = 2*l1+1, d2 = 2*l2+1, d3 = 2*l3+1;
    cdbl q1[5][5], q2[5][5], q3[5][5];
    build_q(l1, q1); build_q(l2, q2); build_q(l3, q3);
    double Wr[5][5][5], Wi[5][5][5];
    double nr = 0.0, ni = 0.0;
    for (int j = 0; j < d1; ++j)
      for (int ll = 0; ll < d2; ++ll)
        for (int n = 0; n < d3; ++n) {
          cdbl acc = {0.0, 0.0};
          for (int i = 0; i < d1; ++i)
            for (int k = 0; k < d2; ++k)
              for (int m = 0; m < d3; ++m) {
                double c = su2_cg(l1, l2, l3, i - l1, k - l2, m - l3);
                if (c == 0.0) continue;
                cdbl t = cmul(q1[i][j], q2[k][ll]);
                t = cmul(t, cconj(q3[m][n]));
                acc = cadd(acc, cscl(t, c));
              }
          Wr[j][ll][n] = acc.re; Wi[j][ll][n] = acc.im;
          nr += acc.re*acc.re; ni += acc.im*acc.im;
        }
    bool useR = (nr >= ni);
    double nrm = sqrt(useR ? nr : ni);
    double scl = (nrm > 0.0) ? (pwk[pk_[p]] / nrm) : 0.0;
    for (int j = 0; j < d1; ++j)
      for (int ll = 0; ll < d2; ++ll)
        for (int n = 0; n < d3; ++n)
          w3j_out[off + (j*d2 + ll)*d3 + n] = (float)((useR ? Wr[j][ll][n] : Wi[j][ll][n]) * scl);
    off += d1*d2*d3;
  }
}

// ---------------------------------------------------------------------------
__global__ void zero_kernel(float* __restrict__ p, int n) {
  int i = blockIdx.x * blockDim.x + threadIdx.x;
  if (i < n) p[i] = 0.0f;
}

// ---------------------------------------------------------------------------
// Dense tensor-product helper (shapes are compile-time, fully unrolled)
// ---------------------------------------------------------------------------
template<int DI, int DJ, int DK>
__device__ __forceinline__ void tp(const float* __restrict__ w3,
                                   const float* __restrict__ xi,
                                   const float* __restrict__ yj,
                                   float wp, float* __restrict__ mk) {
  #pragma unroll
  for (int a = 0; a < DI; ++a) {
    #pragma unroll
    for (int b = 0; b < DJ; ++b) {
      float f = xi[a] * yj[b] * wp;
      #pragma unroll
      for (int c = 0; c < DK; ++c)
        mk[c] = fmaf(w3[(a*DJ + b)*DK + c], f, mk[c]);
    }
  }
}

// ---------------------------------------------------------------------------
// Fused edge kernel: geometry + radial MLP (WMMA) + tensor product + scatter
// One block = 16 edges, 256 threads = 8 waves.
// Node-feature gather goes memory->LDS via GLOBAL_LOAD_ASYNC_TO_LDS_B128 and
// overlaps the MLP + WMMA phases; drained with s_wait_asynccnt before use.
// ---------------------------------------------------------------------------
__global__ __launch_bounds__(256) void edge_kernel(
    const float* __restrict__ nodes, const float* __restrict__ pos,
    const int*   __restrict__ src,   const int*   __restrict__ dst,
    const float* __restrict__ w1,    const float* __restrict__ b1,
    const float* __restrict__ w2,    const float* __restrict__ b2,
    const float* __restrict__ w3j,   float* __restrict__ out, int nEdges)
{
  __shared__ _Float16 sh_h[EB][HIDDEN];                  //  2 KB  (f16 activations = WMMA A)
  __shared__ float    sh_rad[EB][RADCOLS];               // 22 KB  (radial weights, post-GEMM)
  __shared__ __align__(16) float sh_x[EB][FEAT];         // 18 KB  (async-gathered node features)
  __shared__ float    sh_y1[EB][3];
  __shared__ float    sh_y2[EB][5];
  __shared__ float    sh_cut[EB];
  __shared__ float    sh_bes[EB][NB];
  __shared__ int      sh_src[EB];
  __shared__ int      sh_dst[EB];
  __shared__ float    sh_w3j[363];

  const int tid = threadIdx.x;
  const int e0  = blockIdx.x * EB;

  // stage CG tensors
  for (int i = tid; i < 363; i += 256) sh_w3j[i] = w3j[i];

  // ---- Phase A: per-edge geometry (16 threads) ----
  if (tid < EB) {
    int e = e0 + tid;
    float cut = 0.0f; int s = 0, d = 0;
    if (e < nEdges) {
      s = src[e]; d = dst[e];
      float vx = pos[s*3+0] - pos[d*3+0];
      float vy = pos[s*3+1] - pos[d*3+1];
      float vz = pos[s*3+2] - pos[d*3+2];
      float dist = sqrtf(vx*vx + vy*vy + vz*vz);
      float inv  = 1.0f / fmaxf(dist, 1e-12f);
      float x = vx*inv, y = vy*inv, z = vz*inv;
      // Bessel basis (BESSEL_END = 5)
      float t  = dist * 0.2f;
      float st = (t > 0.0f) ? t : 1.0f;
      float pref = sqrtf(0.4f) / st;           // sqrt(2/5)/t
      bool inb = (t > 0.0f) && (t < 1.0f);
      #pragma unroll
      for (int n = 0; n < NB; ++n)
        sh_bes[tid][n] = inb ? pref * __sinf((float)(n+1) * 3.14159265358979f * t) : 0.0f;
      // poly cutoff (RCUT = 4), fold in 1/sqrt(AVG_NEIGHBORS) = 0.2
      float u  = dist * 0.25f;
      float u2 = u*u, u5 = u2*u2*u;
      float env = 1.0f - 6.0f*u5 + 5.0f*u5*u;
      cut = (dist < 4.0f) ? env * 0.2f : 0.0f;
      // spherical harmonics
      const float s3 = 1.73205080757f, s15 = 3.87298334621f, s5h = 1.11803398875f;
      sh_y1[tid][0] = s3*y; sh_y1[tid][1] = s3*z; sh_y1[tid][2] = s3*x;
      sh_y2[tid][0] = s15*x*y; sh_y2[tid][1] = s15*y*z;
      sh_y2[tid][2] = s5h*(3.0f*z*z - 1.0f);
      sh_y2[tid][3] = s15*x*z; sh_y2[tid][4] = 0.5f*s15*(x*x - y*y);
    } else {
      #pragma unroll
      for (int n = 0; n < NB; ++n) sh_bes[tid][n] = 0.0f;
      #pragma unroll
      for (int b = 0; b < 3; ++b) sh_y1[tid][b] = 0.0f;
      #pragma unroll
      for (int b = 0; b < 5; ++b) sh_y2[tid][b] = 0.0f;
    }
    sh_cut[tid] = cut; sh_src[tid] = s; sh_dst[tid] = d;
  }
  __syncthreads();

  // ---- Phase A2: async gather of node features, memory -> LDS (B128) ----
  // 16 edges x 288 floats = 1152 x 16-byte chunks; overlaps phases B and C.
  for (int idx = tid; idx < EB * (FEAT/4); idx += 256) {
    int e = idx / (FEAT/4), c = idx - e*(FEAT/4);        // c = 16B chunk index
    const float* gp = nodes + (size_t)sh_src[e] * FEAT + (size_t)c * 4;
    unsigned lds_off = (unsigned)(uintptr_t)(&sh_x[e][c*4]);
    unsigned long long ga = (unsigned long long)(uintptr_t)gp;
    asm volatile("global_load_async_to_lds_b128 %0, %1, off"
                 :: "v"(lds_off), "v"(ga) : "memory");
  }

  // ---- Phase B: radial MLP layer 1, silu, store f16 (WMMA A matrix) ----
  #pragma unroll
  for (int t = 0; t < 4; ++t) {
    int idx = tid + t*256;            // 0..1023 = 16 edges x 64 hidden
    int e = idx >> 6, c = idx & 63;
    float acc = b1[c];
    #pragma unroll
    for (int n = 0; n < NB; ++n) acc = fmaf(sh_bes[e][n], w1[n*HIDDEN + c], acc);
    float hs = acc / (1.0f + __expf(-acc));     // silu
    sh_h[e][c] = (_Float16)hs;
  }
  __syncthreads();

  // ---- Phase C: radial = silu(h) @ w2 + b2, via v_wmma_f32_16x16x32_f16 ----
  const int lane  = tid & 31;
  const int wv    = tid >> 5;
  const int Mrow  = lane & 15;
  const int khalf = (lane >> 4) * 8;   // A layout: lanes 16-31 hold K+8 halves
  const int kb    = (lane >> 4) * 16;  // unused; B loads use khalf: lanes 16-31 hold K+8 rows
  const int Ncol  = lane & 15;
  const int mo    = (lane >> 4) * 8;   // C/D layout: lanes 16-31 hold M+8 rows

  v16h a0, a1;   // K = 0..31 and K = 32..63 fragments (shared by all tiles)
  {
    const _Float16* hp = &sh_h[Mrow][0];
    #pragma unroll
    for (int t = 0; t < 8; ++t) a0[t]     = hp[khalf + t];
    #pragma unroll
    for (int t = 0; t < 8; ++t) a0[8 + t] = hp[16 + khalf + t];
    #pragma unroll
    for (int t = 0; t < 8; ++t) a1[t]     = hp[32 + khalf + t];
    #pragma unroll
    for (int t = 0; t < 8; ++t) a1[8 + t] = hp[48 + khalf + t];
  }

  for (int tile = wv; tile < RADCOLS/16; tile += 8) {   // 22 N-tiles over 8 waves
    int n0 = tile * 16;
    v16h b0v, b1v;
    #pragma unroll
    for (int t = 0; t < 16; ++t)
      b0v[t] = (_Float16)w2[(size_t)(khalf + t + (t & 8))      * RADCOLS + n0 + Ncol];
    #pragma unroll
    for (int t = 0; t < 16; ++t)
      b1v[t] = (_Float16)w2[(size_t)(32 + khalf + t + (t & 8)) * RADCOLS + n0 + Ncol];
    float bias = b2[n0 + Ncol];
    v8f c;
    #pragma unroll
    for (int g = 0; g < 8; ++g) c[g] = bias;   // b2 folded into accumulator
    c = __builtin_amdgcn_wmma_f32_16x16x32_f16(false, a0, false, b0v, (short)0, c, false, false);
    c = __builtin_amdgcn_wmma_f32_16x16x32_f16(false, a1, false, b1v, (short)0, c, false, false);
    #pragma unroll
    for (int g = 0; g < 8; ++g) {
      int Mg = g + mo;
      sh_rad[Mg][n0 + Ncol] = c[g] * sh_cut[Mg];   // cutoff * 1/sqrt(25) folded in
    }
  }

  // Drain this wave's async LDS writes, then block-wide barrier: after the
  // barrier every wave's gather (and all sh_rad stores) are visible.
  asm volatile("s_wait_asynccnt 0x0" ::: "memory");
  __syncthreads();

  // ---- Phase D: equivariant tensor product + atomic scatter ----
  #pragma unroll
  for (int rep = 0; rep < 2; ++rep) {
    int pidx = tid + rep*256;          // 512 (edge, u) pairs
    int e = pidx >> 5, u = pidx & 31;  // wave-uniform e -> no intra-wave divergence
    if ((e0 + e) < nEdges && sh_cut[e] != 0.0f) {
      float x0a[1] = { sh_x[e][u] };
      float x1v[3], x2v[5], y1v[3], y2v[5];
      #pragma unroll
      for (int a = 0; a < 3; ++a) { x1v[a] = sh_x[e][FDIM + u*3 + a];   y1v[a] = sh_y1[e][a]; }
      #pragma unroll
      for (int a = 0; a < 5; ++a) { x2v[a] = sh_x[e][4*FDIM + u*5 + a]; y2v[a] = sh_y2[e][a]; }
      float y0a[1] = { 1.0f };
      float wr[NPATHS];
      #pragma unroll
      for (int q = 0; q < NPATHS; ++q) wr[q] = sh_rad[e][q*FDIM + u];

      float m0[1] = {0.0f};
      float m1[3] = {0.0f, 0.0f, 0.0f};
      float m2[5] = {0.0f, 0.0f, 0.0f, 0.0f, 0.0f};

      tp<1,1,1>(sh_w3j + 0,   x0a, y0a, wr[0],  m0);
      tp<1,3,3>(sh_w3j + 1,   x0a, y1v, wr[1],  m1);
      tp<1,5,5>(sh_w3j + 10,  x0a, y2v, wr[2],  m2);
      tp<3,1,3>(sh_w3j + 35,  x1v, y0a, wr[3],  m1);
      tp<3,3,1>(sh_w3j + 44,  x1v, y1v, wr[4],  m0);
      tp<3,3,5>(sh_w3j + 53,  x1v, y1v, wr[5],  m2);
      tp<3,5,3>(sh_w3j + 98,  x1v, y2v, wr[6],  m1);
      tp<5,1,5>(sh_w3j + 143, x2v, y0a, wr[7],  m2);
      tp<5,3,3>(sh_w3j + 168, x2v, y1v, wr[8],  m1);
      tp<5,5,1>(sh_w3j + 213, x2v, y2v, wr[9],  m0);
      tp<5,5,5>(sh_w3j + 238, x2v, y2v, wr[10], m2);

      float* op = out + (size_t)sh_dst[e] * FEAT;
      atomicAdd(op + u, m0[0]);
      #pragma unroll
      for (int a = 0; a < 3; ++a) atomicAdd(op + FDIM   + u*3 + a, m1[a]);
      #pragma unroll
      for (int a = 0; a < 5; ++a) atomicAdd(op + 4*FDIM + u*5 + a, m2[a]);
    }
  }
}

// ---------------------------------------------------------------------------
extern "C" void kernel_launch(void* const* d_in, const int* in_sizes, int n_in,
                              void* d_out, int out_size, void* d_ws, size_t ws_size,
                              hipStream_t stream) {
  const float* nodes = (const float*)d_in[0];
  const float* pos   = (const float*)d_in[1];
  const int*   src   = (const int*)  d_in[2];
  const int*   dst   = (const int*)  d_in[3];
  const float* w1    = (const float*)d_in[4];
  const float* b1    = (const float*)d_in[5];
  const float* w2    = (const float*)d_in[6];
  const float* b2    = (const float*)d_in[7];
  float* out = (float*)d_out;
  float* w3j = (float*)d_ws;            // 363 floats of scratch
  int nE = in_sizes[2];

  w3j_init_kernel<<<1, 1, 0, stream>>>(w3j);
  zero_kernel<<<(out_size + 255)/256, 256, 0, stream>>>(out, out_size);
  int nb = (nE + EB - 1) / EB;
  edge_kernel<<<nb, 256, 0, stream>>>(nodes, pos, src, dst, w1, b1, w2, b2, w3j, out, nE);
}
